// MultiHeadAttention_6511170421145
// MI455X (gfx1250) — matrix operand order repaired and checked by reading, not compiled, not run on hardware
//
#include <hip/hip_runtime.h>
#include <hip/hip_bf16.h>

// ---- problem constants (match reference) ----
#define HH   8
#define DD   256
#define DKK  32
#define EE   256
#define BB   16
#define GG   513
#define NPP  256
#define NORMF 0.17677669529663687f   // 1/sqrt(32)

typedef __attribute__((ext_vector_type(16))) __bf16 bf16x16;
typedef __attribute__((ext_vector_type(8)))  __bf16 bf16x8;
typedef __attribute__((ext_vector_type(8)))  float  f32x8;
typedef __attribute__((ext_vector_type(4)))  float  f32x4;
typedef __attribute__((ext_vector_type(4)))  unsigned int u32x4;
typedef __attribute__((ext_vector_type(8)))  int i32x8;
typedef __attribute__((ext_vector_type(4)))  int i32x4;

static __device__ inline f32x8 wmma_bf16(bf16x16 a, bf16x16 b, f32x8 c) {
  // D = A(16x32) * B(32x16) + C, f32 accumulate (v_wmma_f32_16x16x32_bf16)
  return __builtin_amdgcn_wmma_f32_16x16x32_bf16(false, a, false, b, (short)0, c,
                                                 false, false);
}

// A-fragment (16x32 bf16, wave32): lane holds row M=lane%16; K elements are one
// contiguous 16-element run at 16*half.
static __device__ inline bf16x16 packA_f32(const float* __restrict__ row, int hv) {
  const f32x4 a0 = *(const f32x4*)(row + 16 * hv);
  const f32x4 a1 = *(const f32x4*)(row + 16 * hv + 4);
  const f32x4 a2 = *(const f32x4*)(row + 16 * hv + 8);
  const f32x4 a3 = *(const f32x4*)(row + 16 * hv + 12);
  bf16x16 r;
#pragma unroll
  for (int i = 0; i < 4; ++i) {
    r[i]      = (__bf16)a0[i];
    r[4 + i]  = (__bf16)a1[i];
    r[8 + i]  = (__bf16)a2[i];
    r[12 + i] = (__bf16)a3[i];
  }
  return r;
}

static __device__ inline bf16x16 packA_bf16(const __bf16* __restrict__ row, int hv) {
  const bf16x8 lo = *(const bf16x8*)(row + 16 * hv);
  const bf16x8 hi = *(const bf16x8*)(row + 16 * hv + 8);
  bf16x16 r;
#pragma unroll
  for (int i = 0; i < 8; ++i) { r[i] = lo[i]; r[8 + i] = hi[i]; }
  return r;
}

// ------------------------------------------------------------------
// Kernel 1: projection GEMM  Out[h][b][r][dk] (bf16) = X[b, rowOff+r, :] @ W[h]
// ------------------------------------------------------------------
__global__ __launch_bounds__(256) void proj_kernel(const float* __restrict__ X,
                                                   const float* __restrict__ W,
                                                   __bf16* __restrict__ Out,
                                                   int RPB, int rowOff) {
  __shared__ __bf16 sWt[DKK * DD];  // W^T : [dk][d], 16 KB bf16
  const int head = blockIdx.y;
  const int t = threadIdx.x;
  {  // stage W^T (256 threads, one d-row each)
    const float* wrow = W + ((size_t)head * DD + t) * DKK;
    f32x4 w[8];
#pragma unroll
    for (int i = 0; i < 8; ++i) w[i] = *(const f32x4*)(wrow + i * 4);
#pragma unroll
    for (int k = 0; k < DKK; ++k) sWt[k * DD + t] = (__bf16)w[k >> 2][k & 3];
  }
  __syncthreads();

  const int wv = t >> 5, lane = t & 31, lane16 = lane & 15, hv = lane >> 4;
  const int tile = blockIdx.x * 8 + wv;
  const int totalRows = BB * RPB;
  if (tile * 16 >= totalRows) return;  // after the only barrier
  const int m0 = tile * 16;

  const int Rr = m0 + lane16;
  const int bidx = Rr / RPB, ridx = Rr % RPB;
  const float* xrow = X + ((size_t)bidx * GG + rowOff + ridx) * DD;

  f32x8 acc0 = {}, acc1 = {};
#pragma unroll
  for (int k0 = 0; k0 < DD; k0 += 32) {
    bf16x16 a  = packA_f32(xrow + k0, hv);
    bf16x16 b0 = *(const bf16x16*)(&sWt[(lane16)*DD + k0 + 16 * hv]);
    bf16x16 b1 = *(const bf16x16*)(&sWt[(16 + lane16) * DD + k0 + 16 * hv]);
    acc0 = wmma_bf16(a, b0, acc0);
    acc1 = wmma_bf16(a, b1, acc1);
  }
#pragma unroll
  for (int j = 0; j < 8; ++j) {
    const int Rj = m0 + 8 * hv + j;
    const int bj = Rj / RPB, rj = Rj % RPB;
    const size_t o = ((size_t)(head * BB + bj) * RPB + rj) * DKK;
    Out[o + lane16]      = (__bf16)acc0[j];
    Out[o + 16 + lane16] = (__bf16)acc1[j];
  }
}

// ------------------------------------------------------------------
// Kernel 2: fused multi-segment flash attention for one (h, b).
// Whole bf16 K slab (513x32, 32.8 KB) is DMA'd into LDS once by the
// Tensor Data Mover; V is staged per 32-key chunk (transposed).
// ------------------------------------------------------------------
__device__ inline void single_col_term(int m0, int lane16, int hv, int shift,
                                       int nBase, const __bf16* __restrict__ qrows,
                                       const __bf16* __restrict__ Kb,
                                       const __bf16* __restrict__ Vb,
                                       float mrow[8], float lrow[8],
                                       f32x8& acc0, f32x8& acc1) {
  const float NEG_INF = -__builtin_inff();
  float sval = NEG_INF;
  const int qr = m0 + lane16 - shift;
  if (qr >= 0 && qr < NPP) {
    const __bf16* qp = qrows + (size_t)qr * DKK;
    const __bf16* kp = Kb + (size_t)(nBase + qr) * DKK;
    float s = 0.f;
#pragma unroll
    for (int i = 0; i < DKK; ++i) s += (float)qp[i] * (float)kp[i];
    sval = s * NORMF;
  }
#pragma unroll
  for (int j = 0; j < 8; ++j) {
    const float s = __shfl(sval, 8 * hv + j, 16);
    const int qj = m0 + 8 * hv + j;
    const int qrj = qj - shift;
    if (qrj < 0 || qrj >= NPP) continue;  // row not in this region -> -inf col
    const float mnew = fmaxf(mrow[j], s);
    const float scale = (mrow[j] == NEG_INF) ? 0.f : __expf(mrow[j] - mnew);
    const float p = __expf(s - mnew);
    lrow[j] = lrow[j] * scale + p;
    mrow[j] = mnew;
    const __bf16* vp = Vb + (size_t)(nBase + qrj) * DKK;
    acc0[j] = acc0[j] * scale + p * (float)vp[lane16];
    acc1[j] = acc1[j] * scale + p * (float)vp[16 + lane16];
  }
}

#define KPAD 544  // 513 keys padded to 17 chunks of 32

__global__ __launch_bounds__(128) void attn_kernel(
    const __bf16* __restrict__ Q, const __bf16* __restrict__ Kt,
    const __bf16* __restrict__ Vt, const __bf16* __restrict__ Qp1,
    const __bf16* __restrict__ Qp2, const __bf16* __restrict__ Qp3,
    const __bf16* __restrict__ Qd4, const __bf16* __restrict__ Qd5,
    const __bf16* __restrict__ Qd6, float* __restrict__ HeadsOut) {
  __shared__ __bf16 sKall[KPAD * DKK];  // whole K slab [key][dk], 34.8 KB
  __shared__ __bf16 sV[32 * 32];        // V chunk [dk][key] (transposed), 2 KB
  __shared__ __bf16 sP[4 * 16 * 32];    // per-wave probability tiles [row][col]

  const float NEG_INF = -__builtin_inff();
  const int t = threadIdx.x;
  const int b = blockIdx.y, head = blockIdx.z;
  const int wv = t >> 5, lane = t & 31, lane16 = lane & 15, hv = lane >> 4;
  const int tile = blockIdx.x * 4 + wv;
  const bool active = tile < 33;  // ceil(513/16)
  const int m0 = tile * 16;
  const size_t kvO = (size_t)(head * BB + b) * GG * DKK;
  const __bf16* Kb = Kt + kvO;
  const __bf16* Vb = Vt + kvO;
  __bf16* myP = sP + wv * 16 * 32;

  {  // zero the padded tail keys (513..543)
    const int idx = GG * DKK + t * 8;
    if (idx < KPAD * DKK) { bf16x8 z = {}; *(bf16x8*)(&sKall[idx]) = z; }
  }
#if __has_builtin(__builtin_amdgcn_tensor_load_to_lds)
  if (wv == 0) {
    // Tensor DMA descriptor (ISA 8.3-8.6): 1-D tensor of 8208 dwords
    // (513 keys x 64 B of bf16), single tile covering the slab.
    const unsigned long long ga = (unsigned long long)(size_t)Kb;
    const unsigned int lds = (unsigned int)(size_t)(&sKall[0]);  // addr[31:0] = LDS offset
    const int NW = GG * DKK / 2;  // 8208 dwords (data_size = 4 B)
    u32x4 g0;
    g0[0] = 1u;                                   // count=1, user descriptor
    g0[1] = lds;                                  // lds_addr
    g0[2] = (unsigned int)(ga & 0xFFFFFFFFu);     // global_addr[31:0]
    g0[3] = (unsigned int)((ga >> 32) & 0x01FFFFFFu) | (2u << 30);  // addr[56:32], type=2
    i32x8 g1;
    g1[0] = (2 << 16);                            // workgroup_mask=0, data_size=4B
    g1[1] = (NW & 0xFFFF) << 16;                  // tensor_dim0[15:0]
    g1[2] = ((NW >> 16) & 0xFFFF) | (1 << 16);    // tensor_dim0[31:16], tensor_dim1=1
    g1[3] = (NW << 16);                           // tile_dim0 = 8208
    g1[4] = 0;                                    // tile_dim1/2 unused
    g1[5] = NW;                                   // tensor_dim0_stride
    g1[6] = 0;
    g1[7] = 0;
    i32x4 g2 = {0, 0, 0, 0};
    i32x4 g3 = {0, 0, 0, 0};
    i32x8 g4 = {0, 0, 0, 0, 0, 0, 0, 0};
    __builtin_amdgcn_tensor_load_to_lds(g0, g1, g2, g3, g4, 0);
#if __has_builtin(__builtin_amdgcn_s_wait_tensorcnt)
    __builtin_amdgcn_s_wait_tensorcnt(0);
#endif
  }
#else
  // fallback: cooperative vector copy of the K slab
  for (int i = t; i * 8 < GG * DKK; i += 128)
    *(bf16x8*)(&sKall[i * 8]) = *(const bf16x8*)(Kb + i * 8);
#endif
  __syncthreads();

  float mrow[8], lrow[8];
  f32x8 acc0 = {}, acc1 = {};
#pragma unroll
  for (int j = 0; j < 8; ++j) { mrow[j] = NEG_INF; lrow[j] = 0.f; }

  const __bf16* segQ[5]  = {Q, Qp2, Qp3, Qd5, Qd6};
  const int segRows[5]   = {GG, NPP, NPP, NPP, NPP};
  const int segShift[5]  = {0, 1, 1, 257, 257};
  const int segKeyOff[5] = {0, 1, 257, 257, 1};
  const int segNcols[5]  = {GG, NPP, NPP, NPP, NPP};

  for (int seg = 0; seg < 5; ++seg) {
    const int ncols = segNcols[seg], shift = segShift[seg];
    const int rows = segRows[seg], koff = segKeyOff[seg];
    const __bf16* qb = segQ[seg] + (size_t)(head * BB + b) * rows * DKK;

    bf16x16 aQ = {};
    {
      const int qr = m0 + lane16 - shift;
      if (active && qr >= 0 && qr < rows) aQ = packA_bf16(qb + (size_t)qr * DKK, hv);
    }
    const int nch = (ncols + 31) >> 5;
    for (int c = 0; c < nch; ++c) {
      __syncthreads();
      {  // stage V chunk transposed: 32 keys x 32 dk bf16 (2 KB)
        const int krow = t >> 2, dk0 = (t & 3) * 8;
        const int nloc = c * 32 + krow;
        bf16x8 vv = {};
        if (nloc < ncols) {
          vv = *(const bf16x8*)(Vb + (size_t)(koff + nloc) * DKK + dk0);
          if (nloc + 32 < ncols)  // global_prefetch_b8 for next chunk
            __builtin_prefetch(Vb + (size_t)(koff + nloc + 32) * DKK + dk0, 0, 1);
        }
#pragma unroll
        for (int i = 0; i < 8; ++i) sV[(dk0 + i) * 32 + krow] = vv[i];
      }
      __syncthreads();
      if (!active) continue;

      // scores: S(16x32) = Q(16x32dk) x K^T; K B-frags straight from resident slab
      const bf16x16 bk0 =
          *(const bf16x16*)(&sKall[(size_t)(koff + c * 32 + lane16) * DKK + 16 * hv]);
      const bf16x16 bk1 =
          *(const bf16x16*)(&sKall[(size_t)(koff + c * 32 + 16 + lane16) * DKK + 16 * hv]);
      f32x8 z = {};
      const f32x8 s0 = wmma_bf16(aQ, bk0, z);
      const f32x8 s1 = wmma_bf16(aQ, bk1, z);

#pragma unroll
      for (int j = 0; j < 8; ++j) {
        const int qloc = 8 * hv + j;
        const int q = m0 + qloc;
        const int qr = q - shift;
        const bool rv = (qr >= 0) && (qr < rows) && (q < GG);
        const float sc0 = s0[j] * NORMF, sc1 = s1[j] * NORMF;
        const bool v0ok = rv && (c * 32 + lane16 < ncols);
        const bool v1ok = rv && (c * 32 + 16 + lane16 < ncols);
        float ml = fmaxf(v0ok ? sc0 : NEG_INF, v1ok ? sc1 : NEG_INF);
        ml = fmaxf(ml, __shfl_xor(ml, 1, 32));
        ml = fmaxf(ml, __shfl_xor(ml, 2, 32));
        ml = fmaxf(ml, __shfl_xor(ml, 4, 32));
        ml = fmaxf(ml, __shfl_xor(ml, 8, 32));
        const float mnew = fmaxf(mrow[j], ml);
        float scale, p0, p1;
        if (mnew == NEG_INF) {
          scale = 1.f; p0 = 0.f; p1 = 0.f;
        } else {
          scale = (mrow[j] == NEG_INF) ? 0.f : __expf(mrow[j] - mnew);
          p0 = v0ok ? __expf(sc0 - mnew) : 0.f;
          p1 = v1ok ? __expf(sc1 - mnew) : 0.f;
        }
        float rs = p0 + p1;
        rs += __shfl_xor(rs, 1, 32);
        rs += __shfl_xor(rs, 2, 32);
        rs += __shfl_xor(rs, 4, 32);
        rs += __shfl_xor(rs, 8, 32);
        lrow[j] = lrow[j] * scale + rs;
        mrow[j] = mnew;
        acc0[j] *= scale;
        acc1[j] *= scale;
        myP[qloc * 32 + lane16]      = (__bf16)p0;
        myP[qloc * 32 + 16 + lane16] = (__bf16)p1;
      }
      // P: C-layout -> A-layout via per-wave LDS tile (same-wave LDS is in-order)
      const bf16x8 plo = *(const bf16x8*)(&myP[lane16 * 32 + 16 * hv]);
      const bf16x8 phi = *(const bf16x8*)(&myP[lane16 * 32 + 16 * hv + 8]);
      bf16x16 pa;
#pragma unroll
      for (int i = 0; i < 8; ++i) { pa[i] = plo[i]; pa[8 + i] = phi[i]; }
      const bf16x16 bv0 = *(const bf16x16*)(&sV[lane16 * 32 + 16 * hv]);
      const bf16x16 bv1 = *(const bf16x16*)(&sV[(16 + lane16) * 32 + 16 * hv]);
      acc0 = wmma_bf16(pa, bv0, acc0);
      acc1 = wmma_bf16(pa, bv1, acc1);
    }
  }

  if (active) {
    const size_t hb = (size_t)(head * BB + b);
    // diagonal term c_pd: pick row q uses Qp1[q-1] . K[b, 257+(q-1)]
    single_col_term(m0, lane16, hv, 1, 257, Qp1 + hb * NPP * DKK, Kb, Vb,
                    mrow, lrow, acc0, acc1);
    // diagonal term c_dp: delivery row q uses Qd4[q-257] . K[b, 1+(q-257)]
    single_col_term(m0, lane16, hv, 257, 1, Qd4 + hb * NPP * DKK, Kb, Vb,
                    mrow, lrow, acc0, acc1);

#pragma unroll
    for (int j = 0; j < 8; ++j) {
      const int qj = m0 + 8 * hv + j;
      if (qj >= GG) continue;
      const float inv = lrow[j] > 0.f ? 1.f / lrow[j] : 0.f;
      const size_t o = ((size_t)((size_t)b * GG + qj) * HH + head) * DKK;
      HeadsOut[o + lane16]      = acc0[j] * inv;
      HeadsOut[o + 16 + lane16] = acc1[j] * inv;
    }
  }
}

// ------------------------------------------------------------------
// Kernel 3: output projection  out[R][e] = heads[R][:256] @ Wout(256x256)
// ------------------------------------------------------------------
__global__ __launch_bounds__(256) void outproj_kernel(const float* __restrict__ Heads,
                                                      const float* __restrict__ Wout,
                                                      float* __restrict__ Out) {
  __shared__ __bf16 sWt[64 * DD];  // Wout^T slice: [e_local][k], 32 KB
  const int t = threadIdx.x;
  const int e0 = blockIdx.y * 64;
  {
    const float* wrow = Wout + (size_t)t * EE + e0;  // Wout viewed (256 x 256)
#pragma unroll
    for (int i = 0; i < 16; ++i) {
      const f32x4 w = *(const f32x4*)(wrow + i * 4);
#pragma unroll
      for (int p = 0; p < 4; ++p) sWt[(i * 4 + p) * DD + t] = (__bf16)w[p];
    }
  }
  __syncthreads();

  const int wv = t >> 5, lane = t & 31, lane16 = lane & 15, hv = lane >> 4;
  const int tile = blockIdx.x * 8 + wv;
  if (tile >= 513) return;
  const int m0 = tile * 16;
  const float* xrow = Heads + (size_t)(m0 + lane16) * 256;

  f32x8 acc[4] = {};
#pragma unroll
  for (int k0 = 0; k0 < 256; k0 += 32) {
    const bf16x16 a = packA_f32(xrow + k0, hv);
#pragma unroll
    for (int nt = 0; nt < 4; ++nt) {
      const bf16x16 bb =
          *(const bf16x16*)(&sWt[(nt * 16 + lane16) * DD + k0 + 16 * hv]);
      acc[nt] = wmma_bf16(a, bb, acc[nt]);
    }
  }
#pragma unroll
  for (int j = 0; j < 8; ++j) {
    const size_t ro = (size_t)(m0 + 8 * hv + j) * EE + e0;
#pragma unroll
    for (int nt = 0; nt < 4; ++nt) Out[ro + nt * 16 + lane16] = acc[nt][j];
  }
}

// ------------------------------------------------------------------
extern "C" void kernel_launch(void* const* d_in, const int* in_sizes, int n_in,
                              void* d_out, int out_size, void* d_ws, size_t ws_size,
                              hipStream_t stream) {
  const float* q   = (const float*)d_in[0];
  const float* h   = (const float*)d_in[1];
  const float* Wq  = (const float*)d_in[2];
  const float* Wk  = (const float*)d_in[3];
  const float* Wv  = (const float*)d_in[4];
  const float* W1  = (const float*)d_in[5];
  const float* W2  = (const float*)d_in[6];
  const float* W3  = (const float*)d_in[7];
  const float* W4  = (const float*)d_in[8];
  const float* W5  = (const float*)d_in[9];
  const float* W6  = (const float*)d_in[10];
  const float* Wout = (const float*)d_in[11];

  const size_t SZ_QKV = (size_t)HH * BB * GG * DKK;   // 2,101,248 elements
  const size_t SZ_QP  = (size_t)HH * BB * NPP * DKK;  // 1,048,576 elements
  __bf16* wsQ = (__bf16*)d_ws;
  __bf16* wsK = wsQ + SZ_QKV;
  __bf16* wsV = wsK + SZ_QKV;
  __bf16* p1 = wsV + SZ_QKV;
  __bf16* p2 = p1 + SZ_QP;
  __bf16* p3 = p2 + SZ_QP;
  __bf16* p4 = p3 + SZ_QP;
  __bf16* p5 = p4 + SZ_QP;
  __bf16* p6 = p5 + SZ_QP;
  float* wsHeads = (float*)(p6 + SZ_QP);  // [b*G+n][h*32+v], f32 (byte offset is 4-aligned)

  // 9 projection GEMMs (G-row ones: 513 tiles / 8 waves -> 65 blocks; NP: 32 blocks)
  proj_kernel<<<dim3(65, HH), 256, 0, stream>>>(q, Wq, wsQ, GG, 0);
  proj_kernel<<<dim3(65, HH), 256, 0, stream>>>(h, Wk, wsK, GG, 0);
  proj_kernel<<<dim3(65, HH), 256, 0, stream>>>(h, Wv, wsV, GG, 0);
  proj_kernel<<<dim3(32, HH), 256, 0, stream>>>(h, W1, p1, NPP, 1);
  proj_kernel<<<dim3(32, HH), 256, 0, stream>>>(h, W2, p2, NPP, 1);
  proj_kernel<<<dim3(32, HH), 256, 0, stream>>>(h, W3, p3, NPP, 1);
  proj_kernel<<<dim3(32, HH), 256, 0, stream>>>(h, W4, p4, NPP, 257);
  proj_kernel<<<dim3(32, HH), 256, 0, stream>>>(h, W5, p5, NPP, 257);
  proj_kernel<<<dim3(32, HH), 256, 0, stream>>>(h, W6, p6, NPP, 257);

  // fused flash attention over all 7 score regions (K slab via Tensor Data Mover)
  attn_kernel<<<dim3(9, BB, HH), 128, 0, stream>>>(wsQ, wsK, wsV, p1, p2, p3, p4,
                                                   p5, p6, wsHeads);

  // output projection (8208 x 256) @ (256 x 256)
  outproj_kernel<<<dim3(65, 4), 256, 0, stream>>>(wsHeads, Wout, (float*)d_out);
}
